// Quant1Linear_37726992728396
// MI455X (gfx1250) — matrix operand order repaired and checked by reading, not compiled, not run on hardware
//
#include <hip/hip_runtime.h>

typedef __attribute__((ext_vector_type(16))) _Float16 v16h;
typedef __attribute__((ext_vector_type(8)))  float    v8f;

#define IN_FEATURES  8192
#define OUT_FEATURES 8192
#define WORDS        (IN_FEATURES / 32)   // 256 packed qweight rows
#define BLOCK        256                  // 8 waves (wave32)
#define WAVES        8
#define OUTS_PER_BLOCK 32                 // two 16-wide N tiles per wave

__global__ __launch_bounds__(BLOCK)
void q1gemv_wmma(const float* __restrict__ x,
                 const int*   __restrict__ qw,
                 const float* __restrict__ scales,
                 const float* __restrict__ zeros,
                 const float* __restrict__ bias,
                 float*       __restrict__ out)
{
    // x staged as f16 (16 KB) + 128 B zero region for non-row A lanes
    __shared__ alignas(16) _Float16 xh[IN_FEATURES + 64];
    // byte -> 8 packed f16 {0,1} lookup table (4 KB)
    __shared__ alignas(16) uint4 lut[256];
    __shared__ float ypart[OUTS_PER_BLOCK];
    __shared__ float xsum_s;

    const int t    = threadIdx.x;
    const int lane = t & 31;
    const int wid  = t >> 5;
    const int o0   = blockIdx.x * OUTS_PER_BLOCK;

    // ---- phase 0: zero-init + LUT build ----
    if (t < OUTS_PER_BLOCK) ypart[t] = 0.0f;
    if (t == 0)             xsum_s   = 0.0f;
    {   // zero the A-zero region (halves 8192..8255)
        unsigned int* xw = reinterpret_cast<unsigned int*>(xh);
        if (t < 32) xw[IN_FEATURES / 2 + t] = 0u;
    }
    {   // one LUT entry per thread (BLOCK == 256)
        const unsigned int e = (unsigned int)t;
        uint4 v;
        v.x = ((e &   1u) ? 0x3C00u : 0u) | ((e &   2u) ? 0x3C000000u : 0u);
        v.y = ((e &   4u) ? 0x3C00u : 0u) | ((e &   8u) ? 0x3C000000u : 0u);
        v.z = ((e &  16u) ? 0x3C00u : 0u) | ((e &  32u) ? 0x3C000000u : 0u);
        v.w = ((e &  64u) ? 0x3C00u : 0u) | ((e & 128u) ? 0x3C000000u : 0u);
        lut[t] = v;
    }
    __syncthreads();

    // ---- phase 1: stage x -> LDS f16 (vectorized), reduce xsum in f32 ----
    float lsum = 0.0f;
    {
        const float4* x4  = reinterpret_cast<const float4*>(x);
        uint2*        xh2 = reinterpret_cast<uint2*>(xh);
#pragma unroll
        for (int j = 0; j < IN_FEATURES / 4 / BLOCK; ++j) {   // 8 iterations
            const int idx = t + BLOCK * j;
            const float4 v = x4[idx];                         // global_load_b128
            lsum += (v.x + v.y) + (v.z + v.w);
            union { _Float16 h[4]; uint2 u2; } p;
            p.h[0] = (_Float16)v.x;  p.h[1] = (_Float16)v.y;  // v_cvt_pk_f16_f32
            p.h[2] = (_Float16)v.z;  p.h[3] = (_Float16)v.w;
            xh2[idx] = p.u2;                                  // ds_store_b64
        }
    }
    atomicAdd(&xsum_s, lsum);   // ds_add_f32
    __syncthreads();

    // ---- phase 2: per-wave K-slice over two 16-output tiles ----
    const int          nsub    = lane & 15;      // N column within tile
    const unsigned int shift   = (lane & 16);    // K-half of 32-bit word
    const bool         rowlane = (nsub == 0);    // lanes 0,16 carry A row M=0
    const int          g       = lane >> 4;      // A K-interleave group

    // A walk: row lanes step through xh; others sit on the zero region.
    const uint4* ap   = reinterpret_cast<const uint4*>(xh) +
                        (rowlane ? (wid * 4 + 2 * g) : (IN_FEATURES / 8));
    const int    ainc = rowlane ? (WAVES * 4) : 0;   // uint4 per iteration

    const int* qp = qw + wid * OUT_FEATURES + o0 + nsub;

    v8f c0 = {0.f,0.f,0.f,0.f,0.f,0.f,0.f,0.f};
    v8f c1 = {0.f,0.f,0.f,0.f,0.f,0.f,0.f,0.f};

#pragma unroll 4
    for (int i = 0; i < WORDS / WAVES; ++i) {
        // qweight is single-use streaming data: non-temporal loads
        const unsigned int q0 = (unsigned int)__builtin_nontemporal_load(qp);
        const unsigned int q1 = (unsigned int)__builtin_nontemporal_load(qp + 16);
        __builtin_prefetch((const void*)(qp + 2 * WAVES * OUT_FEATURES), 0, 0);
        qp += WAVES * OUT_FEATURES;

        // A (16x32 f16): row 0 = x chunk, rows 1..15 zero via zero region
        union { uint4 v2[2]; v16h h; } a;
        a.v2[0] = ap[0];          // ds_load_b128
        a.v2[1] = ap[1];          // ds_load_b128 offset:16
        ap += ainc;

        // B tiles (32x16 f16) via byte LUT: 2 ds_load_b128 each
        union { uint4 v2[2]; v16h h; } b0, b1;
        const unsigned int h0 = (q0 >> shift) & 0xFFFFu;
        b0.v2[0] = lut[h0 & 0xFFu];
        b0.v2[1] = lut[h0 >> 8];
        const unsigned int h1 = (q1 >> shift) & 0xFFFFu;
        b1.v2[0] = lut[h1 & 0xFFu];
        b1.v2[1] = lut[h1 >> 8];

        c0 = __builtin_amdgcn_wmma_f32_16x16x32_f16(
                 false, a.h, false, b0.h, (short)0, c0, false, false);
        c1 = __builtin_amdgcn_wmma_f32_16x16x32_f16(
                 false, a.h, false, b1.h, (short)0, c1, false, false);
    }

    // Row M=0 of D lives in VGPR0 of lanes 0..15 (N = lane). Reduce 8 waves.
    if (lane < 16) {
        atomicAdd(&ypart[lane],      c0[0]);
        atomicAdd(&ypart[16 + lane], c1[0]);
    }
    __syncthreads();

    if (t < OUTS_PER_BLOCK) {
        const int o = o0 + t;
        out[o] = ypart[t] * scales[o] - xsum_s * zeros[o] + bias[o];
    }
}

extern "C" void kernel_launch(void* const* d_in, const int* in_sizes, int n_in,
                              void* d_out, int out_size, void* d_ws, size_t ws_size,
                              hipStream_t stream) {
    const float* x      = (const float*)d_in[0];
    const int*   qw     = (const int*)  d_in[1];
    const float* scales = (const float*)d_in[2];
    const float* zeros  = (const float*)d_in[3];
    const float* bias   = (const float*)d_in[4];
    float* out = (float*)d_out;

    dim3 grid(OUT_FEATURES / OUTS_PER_BLOCK);   // 256 blocks
    dim3 block(BLOCK);                           // 256 threads = 8 waves
    q1gemv_wmma<<<grid, block, 0, stream>>>(x, qw, scales, zeros, bias, out);
}
